// Gemma4VisionAttention_86320252715809
// MI455X (gfx1250) — matrix operand order repaired and checked by reading, not compiled, not run on hardware
//
#include <hip/hip_runtime.h>
#include <stdint.h>

// ---------------------------------------------------------------------------
// Gemma vision attention on gfx1250 (MI455X).
// bf16 WMMA (16x16x32, fp32 accum) everywhere; fp32 pointwise math.
// CDNA5-specific paths: v_wmma_f32_16x16x32_bf16, tensor_load_to_lds (TDM)
// double-buffered K/V staging, s_wait_tensorcnt, global_prefetch_b8.
// ---------------------------------------------------------------------------

typedef __attribute__((ext_vector_type(16))) __bf16 v16bf;
typedef __attribute__((ext_vector_type(8)))  __bf16 bf16x8;
typedef __attribute__((ext_vector_type(8)))  float  v8f;
typedef __attribute__((ext_vector_type(4)))  float  f32x4;
typedef __attribute__((ext_vector_type(4)))  unsigned int u32x4;
typedef __attribute__((ext_vector_type(4)))  int    i32x4;
typedef __attribute__((ext_vector_type(8)))  int    i32x8;

#define BB   4
#define SS   2048
#define HID  1152
#define NH   16
#define NKV  8
#define DH   72
#define DPAD 96     // q/k head dim padded to 3 * 32 for K=32 WMMA chunks
#define VDP  80     // transposed-V head dim padded to 5 * 16 output tiles
#define ROWS (BB*SS)          // 8192
#define QKVC (HID + 2*NKV*DH) // 2304

#if defined(__has_builtin)
#if __has_builtin(__builtin_amdgcn_tensor_load_to_lds) && \
    __has_builtin(__builtin_amdgcn_s_wait_tensorcnt)
#define USE_TDM 1
#endif
#endif
#ifndef USE_TDM
#define USE_TDM 0
#endif

static __device__ __forceinline__ v8f wmma_bf16(v16bf a, v16bf b, v8f c) {
  return __builtin_amdgcn_wmma_f32_16x16x32_bf16(false, a, false, b,
                                                 (short)0, c, false, false);
}

#if USE_TDM
// 2D TENSOR_LOAD_TO_LDS: hand-packed D# per ISA 8.3/8.4 (data_size = 2 bytes).
static __device__ __forceinline__ void tdm_load_2d(
    unsigned int lds_off, const void* gptr,
    unsigned int tens_d0, unsigned int tens_d1,
    unsigned int tile_d0, unsigned int tile_d1, unsigned long long stride0) {
  const unsigned long long ga = (unsigned long long)(uintptr_t)gptr;
  u32x4 g0;
  g0[0] = 1u;                                   // count=1, user-mode, no gather
  g0[1] = lds_off;                              // LDS byte address
  g0[2] = (unsigned int)(ga & 0xFFFFFFFFu);     // global_addr[31:0]
  g0[3] = (unsigned int)((ga >> 32) & 0x01FFFFFFu) | (2u << 30);  // [56:32]|type=2
  i32x8 g1;
  g1[0] = (int)0x00010000u;                     // wg_mask=0, data_size=1 (2B)
  g1[1] = (int)((tens_d0 & 0xFFFFu) << 16);     // tensor_dim0[15:0]
  g1[2] = (int)(((tens_d0 >> 16) & 0xFFFFu) | ((tens_d1 & 0xFFFFu) << 16));
  g1[3] = (int)(((tens_d1 >> 16) & 0xFFFFu) | ((tile_d0 & 0xFFFFu) << 16));
  g1[4] = (int)(tile_d1 & 0xFFFFu);             // tile_dim1, tile_dim2=0
  g1[5] = (int)(stride0 & 0xFFFFFFFFu);         // tensor_dim0_stride[31:0]
  g1[6] = (int)((stride0 >> 32) & 0xFFFFu);     // stride[47:32], dim1_stride=0
  g1[7] = 0;
  i32x4 g2 = {0, 0, 0, 0};
  i32x4 g3 = {0, 0, 0, 0};
#if defined(__clang_major__) && (__clang_major__ >= 23)
  i32x8 g4 = {0, 0, 0, 0, 0, 0, 0, 0};
  __builtin_amdgcn_tensor_load_to_lds(g0, g1, g2, g3, g4, 0);
#else
  __builtin_amdgcn_tensor_load_to_lds(g0, g1, g2, g3, 0);
#endif
}
#endif

// ---------------------------------------------------------------------------
// One-time repack kernels (cheap: ~4M elements vs 23.3 TB/s).
// ---------------------------------------------------------------------------
__global__ __launch_bounds__(256) void cvt_hidden_bf16(
    const float* __restrict__ src, __bf16* __restrict__ dst) {
  const int i = (blockIdx.x * 256 + threadIdx.x) * 4;
  f32x4 v = *(const f32x4*)(src + i);
#pragma unroll
  for (int e = 0; e < 4; ++e) dst[i + e] = (__bf16)v[e];
}

// Wcat^T[n][k] (bf16, n in 0..2303) from row-major Wq/Wk/Wv (f32).
__global__ __launch_bounds__(256) void transpose_wcat(
    const float* __restrict__ Wq, const float* __restrict__ Wk,
    const float* __restrict__ Wv, __bf16* __restrict__ dst) {
  const int id = blockIdx.x * 256 + threadIdx.x;   // over QKVC*HID
  const int k = id % HID;
  const int n = id / HID;
  float v;
  if (n < HID)                v = Wq[(size_t)k * HID + n];
  else if (n < HID + NKV*DH)  v = Wk[(size_t)k * (NKV*DH) + (n - HID)];
  else                        v = Wv[(size_t)k * (NKV*DH) + (n - HID - NKV*DH)];
  dst[(size_t)n * HID + k] = (__bf16)v;
}

// Generic W^T[n][k] bf16 from row-major W[k][n] f32 (square K==N==HID).
__global__ __launch_bounds__(256) void transpose_w(
    const float* __restrict__ W, __bf16* __restrict__ dst) {
  const int id = blockIdx.x * 256 + threadIdx.x;
  const int k = id % HID;
  const int n = id / HID;
  dst[(size_t)n * HID + k] = (__bf16)W[(size_t)k * HID + n];
}

// ---------------------------------------------------------------------------
// GEMM: C_f32[M x N] = A_bf16[M x K] * BT_bf16[N x K]^T.
// Wave tile 32(M) x 64(N): 2 A-frags x 4 B-frags -> 8 WMMAs / K=32 step.
// 8 waves/WG arranged 4(M) x 2(N): 128 x 128 per workgroup.
// A frag: lane=row, kbase=lane<16?0:16, 16 contiguous K (two 16B loads).
// B frag: lane=col, kbase=lane<16?0:16, 16 contiguous K -> one 32B load.
// ---------------------------------------------------------------------------
__global__ __launch_bounds__(256) void gemm_bf16t(
    const __bf16* __restrict__ A, const __bf16* __restrict__ BT,
    float* __restrict__ C, int K, int N) {
  const int lane = threadIdx.x & 31;
  const int w    = threadIdx.x >> 5;
  const int m0 = blockIdx.x * 128 + (w & 3) * 32;
  const int n0 = blockIdx.y * 128 + (w >> 2) * 64;
  const int l15 = lane & 15;
  const int hi  = lane >> 4;
  const int ak  = hi ? 16 : 0;
  const int bk  = hi ? 16 : 0;

  const __bf16* Ap[2];
  Ap[0] = A + (size_t)(m0 + l15) * K + ak;
  Ap[1] = Ap[0] + (size_t)16 * K;
  const __bf16* Bp[4];
#pragma unroll
  for (int j = 0; j < 4; ++j)
    Bp[j] = BT + (size_t)(n0 + j * 16 + l15) * K + bk;

  v8f acc[2][4];
#pragma unroll
  for (int i = 0; i < 2; ++i)
#pragma unroll
    for (int j = 0; j < 4; ++j) acc[i][j] = (v8f){};

  for (int k0 = 0; k0 < K; k0 += 32) {
    __builtin_prefetch((const void*)(Ap[0] + k0 + 128), 0, 1);
    __builtin_prefetch((const void*)(Bp[0] + k0 + 128), 0, 1);
    v16bf a[2], b[4];
#pragma unroll
    for (int i = 0; i < 2; ++i) {
      bf16x8 lo = *(const bf16x8*)(Ap[i] + k0);
      bf16x8 hg = *(const bf16x8*)(Ap[i] + k0 + 8);
#pragma unroll
      for (int e = 0; e < 8; ++e) { a[i][e] = lo[e]; a[i][8 + e] = hg[e]; }
    }
#pragma unroll
    for (int j = 0; j < 4; ++j) b[j] = *(const v16bf*)(Bp[j] + k0);
#pragma unroll
    for (int i = 0; i < 2; ++i)
#pragma unroll
      for (int j = 0; j < 4; ++j) acc[i][j] = wmma_bf16(a[i], b[j], acc[i][j]);
  }
  // C layout: vgpr r, lane -> row = r + 8*hi, col = l15
#pragma unroll
  for (int i = 0; i < 2; ++i) {
#pragma unroll
    for (int j = 0; j < 4; ++j) {
      float* Cp = C + (size_t)(m0 + i * 16 + 8 * hi) * N + n0 + j * 16 + l15;
#pragma unroll
      for (int r = 0; r < 8; ++r) Cp[(size_t)r * N] = acc[i][j][r];
    }
  }
}

// ---------------------------------------------------------------------------
// Epilogue: per-(row, head) RMSNorm + 2-axis RoPE, f32 -> bf16 repack.
//   slot 0..15 : q head  -> q_ws[b][h][s][DPAD]   (pad 72..95 = 0)
//   slot 16..23: k head  -> k_ws[b][kv][s][DPAD]  (pad 72..95 = 0)
//   slot 24..31: v head  -> vT_ws[b][kv][d][S]    (pad rows 72..79 = 0)
// ---------------------------------------------------------------------------
__global__ __launch_bounds__(256) void qkv_epilogue(
    const float* __restrict__ qkv, const float* __restrict__ cosb,
    const float* __restrict__ sinb, const float* __restrict__ qw,
    const float* __restrict__ kw, __bf16* __restrict__ q_ws,
    __bf16* __restrict__ k_ws, __bf16* __restrict__ vt_ws) {
  const int t = blockIdx.x * blockDim.x + threadIdx.x;
  const int slot = t & 31;
  const int row  = t >> 5;
  if (row >= ROWS) return;
  const int b = row / SS, s = row % SS;

  int col;
  const float* wvec = nullptr;
  bool do_rope = true;
  int head = 0;
  if (slot < 16)      { head = slot;      col = head * DH;               wvec = qw; }
  else if (slot < 24) { head = slot - 16; col = HID + head * DH;         wvec = kw; }
  else                { head = slot - 24; col = HID + NKV*DH + head*DH;  do_rope = false; }

  const float* src = qkv + (size_t)row * QKVC + col;
  float y[DH];
  float ssq = 0.f;
#pragma unroll
  for (int d = 0; d < DH; ++d) { y[d] = src[d]; ssq += y[d] * y[d]; }
  const float rn = rsqrtf(ssq * (1.0f / DH) + 1e-6f);

  if (!do_rope) {
    __bf16* dst = vt_ws + ((size_t)(b * NKV + head) * VDP) * SS + s;
#pragma unroll
    for (int d = 0; d < DH; ++d) dst[(size_t)d * SS] = (__bf16)(y[d] * rn);
#pragma unroll
    for (int d = DH; d < VDP; ++d) dst[(size_t)d * SS] = (__bf16)0.f;
    return;
  }

#pragma unroll
  for (int d = 0; d < DH; ++d) y[d] = y[d] * rn * (1.f + wvec[d]);

  float out[DH];
  const float* cp = cosb + ((size_t)b * SS + s) * DH;
  const float* sp = sinb + ((size_t)b * SS + s) * DH;
#pragma unroll
  for (int i = 0; i < 2; ++i) {
#pragma unroll
    for (int j = 0; j < 36; ++j) {
      const int d = i * 36 + j;
      const float rot = (j < 18) ? -y[d + 18] : y[d - 18];
      out[d] = y[d] * cp[d] + rot * sp[d];
    }
  }

  __bf16* dst = (slot < 16)
      ? q_ws + ((size_t)(b * NH + head) * SS + s) * DPAD
      : k_ws + ((size_t)(b * NKV + head) * SS + s) * DPAD;
#pragma unroll
  for (int d = 0; d < DH; ++d) dst[d] = (__bf16)out[d];
#pragma unroll
  for (int d = DH; d < DPAD; ++d) dst[d] = (__bf16)0.f;
}

// ---------------------------------------------------------------------------
// Flash attention (no 1/sqrt(D) scale, per reference).
// WG = 8 waves = 128 queries of one (b,h). K/V tiles (32 keys) are staged in
// LDS once per WG via the Tensor Data Mover (double-buffered, wave 0 issues,
// s_wait_tensorcnt + barrier), then all waves read WMMA fragments from LDS.
// Per 32-key step: 6 QK^T WMMAs + 5 P.V WMMAs.
// ---------------------------------------------------------------------------
__global__ __launch_bounds__(256) void flash_attn(
    const __bf16* __restrict__ q_ws, const __bf16* __restrict__ k_ws,
    const __bf16* __restrict__ vt_ws, const float* __restrict__ mask,
    __bf16* __restrict__ attn) {
#if USE_TDM
  __shared__ __align__(64) __bf16 kt[2][32][DPAD];  // 2 x 6KB
  __shared__ __align__(64) __bf16 vt[2][VDP][32];   // 2 x 5KB
#else
  __shared__ __align__(64) __bf16 kt[1][32][DPAD];
  __shared__ __align__(64) __bf16 vt[1][VDP][32];
#endif
  __shared__ __align__(64) __bf16 lds_p[8][16][32]; // per-wave P staging, 8KB

  const int lane = threadIdx.x & 31;
  const int w    = threadIdx.x >> 5;
  const int bh   = blockIdx.x >> 4;     // b*NH + h
  const int qblk = blockIdx.x & 15;
  const int b  = bh >> 4;
  const int h  = bh & 15;
  const int hk = h >> 1;                // GQA: 2 q heads per kv head
  const int q0 = qblk * 128 + w * 16;
  const int l15 = lane & 15;
  const int hi  = lane >> 4;
  const int ak  = hi ? 16 : 0;
  const int bk  = hi ? 16 : 0;

  // Q fragments: 3 chunks of K=32 along DPAD (resident for the whole pass)
  const __bf16* qrow = q_ws + ((size_t)bh * SS + (q0 + l15)) * DPAD;
  v16bf aq[3];
#pragma unroll
  for (int c = 0; c < 3; ++c) {
    bf16x8 lo = *(const bf16x8*)(qrow + c * 32 + ak);
    bf16x8 hg = *(const bf16x8*)(qrow + c * 32 + ak + 8);
#pragma unroll
    for (int e = 0; e < 8; ++e) { aq[c][e] = lo[e]; aq[c][8 + e] = hg[e]; }
  }

  const __bf16* kbase = k_ws + ((size_t)(b * NKV + hk) * SS) * DPAD;
  const __bf16* vbase = vt_ws + ((size_t)(b * NKV + hk) * VDP) * SS;
  const float*  mbase = mask + ((size_t)b * SS + q0) * SS;

  float mrow[8], lrow[8];
  v8f o[5];
#pragma unroll
  for (int r = 0; r < 8; ++r) { mrow[r] = -1e30f; lrow[r] = 0.f; }
#pragma unroll
  for (int t = 0; t < 5; ++t) o[t] = (v8f){};

#if USE_TDM
  if (w == 0) {  // prologue: stage tiles for j0 = 0 into buffer 0
    tdm_load_2d((unsigned int)(uintptr_t)&kt[0][0][0], kbase,
                DPAD, SS, DPAD, 32, DPAD);
    tdm_load_2d((unsigned int)(uintptr_t)&vt[0][0][0], vbase,
                SS, VDP, 32, VDP, SS);
  }
#endif

  for (int j0 = 0, it = 0; j0 < SS; j0 += 32, ++it) {
#if USE_TDM
    const int cb = it & 1;
    __syncthreads();  // all waves done reading the buffer we overwrite next
    if (w == 0) {
      if (j0 + 32 < SS) {
        tdm_load_2d((unsigned int)(uintptr_t)&kt[cb ^ 1][0][0],
                    kbase + (size_t)(j0 + 32) * DPAD, DPAD, SS, DPAD, 32, DPAD);
        tdm_load_2d((unsigned int)(uintptr_t)&vt[cb ^ 1][0][0],
                    vbase + (j0 + 32), SS, VDP, 32, VDP, SS);
        __builtin_amdgcn_s_wait_tensorcnt(2);  // current buffer complete
      } else {
        __builtin_amdgcn_s_wait_tensorcnt(0);
      }
    }
    __syncthreads();  // buffer cb visible to all waves
#else
    const int cb = 0;
    __syncthreads();
    for (int idx = threadIdx.x; idx < 32 * DPAD; idx += 256)
      kt[0][idx / DPAD][idx % DPAD] = kbase[(size_t)(j0 + idx / DPAD) * DPAD + idx % DPAD];
    for (int idx = threadIdx.x; idx < VDP * 32; idx += 256)
      vt[0][idx >> 5][idx & 31] = vbase[(size_t)(idx >> 5) * SS + j0 + (idx & 31)];
    __syncthreads();
#endif

    v8f s0 = {}, s1 = {};
#pragma unroll
    for (int c = 0; c < 3; ++c) {
      v16bf b0 = *(const v16bf*)(&kt[cb][l15][c * 32 + bk]);
      v16bf b1 = *(const v16bf*)(&kt[cb][16 + l15][c * 32 + bk]);
      s0 = wmma_bf16(aq[c], b0, s0);
      s1 = wmma_bf16(aq[c], b1, s1);
    }

    float pr0[8], pr1[8];
#pragma unroll
    for (int r = 0; r < 8; ++r) {
      const float* mrp = mbase + (size_t)(r + 8 * hi) * SS + j0 + l15;
      s0[r] += mrp[0];
      s1[r] += mrp[16];
      float mx = fmaxf(s0[r], s1[r]);
      mx = fmaxf(mx, __shfl_xor(mx, 1, 32));
      mx = fmaxf(mx, __shfl_xor(mx, 2, 32));
      mx = fmaxf(mx, __shfl_xor(mx, 4, 32));
      mx = fmaxf(mx, __shfl_xor(mx, 8, 32));
      const float mnew = fmaxf(mrow[r], mx);
      const float scale = __expf(mrow[r] - mnew);
      mrow[r] = mnew;
      pr0[r] = __expf(s0[r] - mnew);
      pr1[r] = __expf(s1[r] - mnew);
      float rs = pr0[r] + pr1[r];
      rs += __shfl_xor(rs, 1, 32);
      rs += __shfl_xor(rs, 2, 32);
      rs += __shfl_xor(rs, 4, 32);
      rs += __shfl_xor(rs, 8, 32);
      lrow[r] = lrow[r] * scale + rs;
#pragma unroll
      for (int t = 0; t < 5; ++t) o[t][r] *= scale;
    }

    // C-layout -> LDS -> A-layout repack of probabilities (bf16)
#pragma unroll
    for (int r = 0; r < 8; ++r) {
      lds_p[w][r + 8 * hi][l15]      = (__bf16)pr0[r];
      lds_p[w][r + 8 * hi][16 + l15] = (__bf16)pr1[r];
    }
    asm volatile("s_wait_dscnt 0" ::: "memory");
    v16bf ap;
    {
      const __bf16* pr = &lds_p[w][l15][0];
      bf16x8 lo = *(const bf16x8*)(pr + ak);
      bf16x8 hg = *(const bf16x8*)(pr + ak + 8);
#pragma unroll
      for (int e = 0; e < 8; ++e) { ap[e] = lo[e]; ap[8 + e] = hg[e]; }
    }
    asm volatile("s_wait_dscnt 0" ::: "memory");  // keep later DS after loads

#pragma unroll
    for (int t = 0; t < 5; ++t) {
      v16bf bv = *(const v16bf*)(&vt[cb][t * 16 + l15][bk]);
      o[t] = wmma_bf16(ap, bv, o[t]);
    }
  }

  // normalize and store bf16 attention output (rows of the Wo GEMM)
#pragma unroll
  for (int r = 0; r < 8; ++r) {
    const float inv = 1.f / lrow[r];
    const int m = r + 8 * hi;
    __bf16* dst = attn + ((size_t)(b * SS) + q0 + m) * HID + h * DH;
#pragma unroll
    for (int t = 0; t < 5; ++t) {
      const int d = t * 16 + l15;
      if (d < DH) dst[d] = (__bf16)(o[t][r] * inv);
    }
  }
}

// ---------------------------------------------------------------------------
extern "C" void kernel_launch(void* const* d_in, const int* in_sizes, int n_in,
                              void* d_out, int out_size, void* d_ws, size_t ws_size,
                              hipStream_t stream) {
  (void)in_sizes; (void)n_in; (void)out_size; (void)ws_size;
  const float* hidden = (const float*)d_in[0];
  const float* cosb   = (const float*)d_in[1];
  const float* sinb   = (const float*)d_in[2];
  const float* mask   = (const float*)d_in[3];
  // d_in[4] = position_ids (unused: cos/sin provided directly)
  const float* Wq = (const float*)d_in[5];
  const float* Wk = (const float*)d_in[6];
  const float* Wv = (const float*)d_in[7];
  const float* Wo = (const float*)d_in[8];
  const float* qw = (const float*)d_in[9];
  const float* kw = (const float*)d_in[10];

  char* ws = (char*)d_ws;
  const size_t qkv_bytes  = (size_t)ROWS * QKVC * sizeof(float);           // 75.5 MB
  const size_t q_bytes    = (size_t)BB * NH  * SS * DPAD * sizeof(__bf16); // 25.2 MB
  const size_t k_bytes    = (size_t)BB * NKV * SS * DPAD * sizeof(__bf16); // 12.6 MB
  const size_t v_bytes    = (size_t)BB * NKV * VDP * SS * sizeof(__bf16);  // 10.5 MB
  const size_t hb_bytes   = (size_t)ROWS * HID * sizeof(__bf16);           // 18.9 MB
  const size_t wcat_bytes = (size_t)QKVC * HID * sizeof(__bf16);           //  5.3 MB

  float*  qkv_raw  = (float*)ws;
  __bf16* attn_ws  = (__bf16*)ws;  // aliases qkv_raw (dead after epilogue)
  __bf16* q_ws     = (__bf16*)(ws + qkv_bytes);
  __bf16* k_ws     = (__bf16*)(ws + qkv_bytes + q_bytes);
  __bf16* vt_ws    = (__bf16*)(ws + qkv_bytes + q_bytes + k_bytes);
  __bf16* hidden_b = (__bf16*)(ws + qkv_bytes + q_bytes + k_bytes + v_bytes);
  __bf16* wcat_t   = (__bf16*)((char*)hidden_b + hb_bytes);
  __bf16* wo_t     = (__bf16*)((char*)wcat_t + wcat_bytes);

  // 0) one-time bf16 repack of activations + transposed weights
  cvt_hidden_bf16<<<(ROWS * HID) / (256 * 4), 256, 0, stream>>>(hidden, hidden_b);
  transpose_wcat<<<(QKVC * HID) / 256, 256, 0, stream>>>(Wq, Wk, Wv, wcat_t);
  transpose_w<<<(HID * HID) / 256, 256, 0, stream>>>(Wo, wo_t);
  // 1) fused QKV projection: single GEMM, N = 2304
  gemm_bf16t<<<dim3(ROWS / 128, QKVC / 128), 256, 0, stream>>>(
      hidden_b, wcat_t, qkv_raw, HID, QKVC);
  // 2) RMSNorm + RoPE + bf16 repack (q/k padded, v transposed)
  qkv_epilogue<<<(ROWS * 32) / 256, 256, 0, stream>>>(qkv_raw, cosb, sinb, qw, kw,
                                                      q_ws, k_ws, vt_ws);
  // 3) flash attention (TDM-staged K/V tiles; writes over dead qkv_raw region)
  flash_attn<<<BB * NH * (SS / 128), 256, 0, stream>>>(q_ws, k_ws, vt_ws, mask,
                                                       attn_ws);
  // 4) output projection -> fp32 d_out
  gemm_bf16t<<<dim3(ROWS / 128, HID / 128), 256, 0, stream>>>(
      attn_ws, wo_t, (float*)d_out, HID, HID);
}
